// GATNet_57191784513627
// MI455X (gfx1250) — matrix operand order repaired and checked by reading, not compiled, not run on hardware
//
#include <hip/hip_runtime.h>
#include <hip/hip_bf16.h>

typedef __attribute__((ext_vector_type(16))) _Float16 v16h;
typedef __attribute__((ext_vector_type(8)))  float    v8f;

#define LRELU_SLOPE 0.2f
#define EPS_A 1e-16f

// ---------- small utilities ----------
__global__ void fill_f32(float* __restrict__ p, float v, long n) {
  long i = (long)blockIdx.x * blockDim.x + threadIdx.x;
  if (i < n) p[i] = v;
}

// mean of edge_attr over E rows -> mean[16]
__global__ void mean_attr_kernel(const float* __restrict__ ea, float* __restrict__ mean,
                                 int E, int T) {
  int t = blockIdx.x * blockDim.x + threadIdx.x;
  if (t >= T) return;
  int dim = t & 15;
  int stride = T >> 4;
  float acc = 0.f;
  for (int r = t >> 4; r < E; r += stride) acc += ea[(size_t)r * 16 + dim];
  atomicAdd(&mean[dim], acc * (1.f / (float)E));
}

// M[d*4+h] = sum_c We[d,64] chunk (h*16+c) * att_e[h,c]   -> collapses edge proj + att dot
__global__ void collapse_We(const float* __restrict__ We, const float* __restrict__ att_e,
                            float* __restrict__ M) {
  int t = threadIdx.x;
  if (t >= 64) return;
  int d = t >> 2, h = t & 3;
  float acc = 0.f;
  #pragma unroll
  for (int c = 0; c < 16; ++c) acc += We[d * 64 + h * 16 + c] * att_e[h * 16 + c];
  M[t] = acc;
}

// ---------- WMMA node GEMM: feat[N,64] = in[N,KDIM] @ W[KDIM,64] ----------
// block = 128 threads = 4 waves; wave w owns output columns [16w, 16w+16).
// blockIdx.x owns 16 rows. Fragment layouts per CDNA5 ISA §7.12.2.
template <int KDIM>
__global__ void node_gemm(const float* __restrict__ in, const float* __restrict__ W,
                          float* __restrict__ feat) {
  const int lane = threadIdx.x & 31;
  const int wave = threadIdx.x >> 5;
  const int half = lane >> 4;       // lane group 0-15 / 16-31
  const int lm   = lane & 15;
  const int row0 = blockIdx.x * 16;
  const int col  = wave * 16 + lm;

  v8f acc = {};
  #pragma unroll
  for (int kc = 0; kc < KDIM; kc += 32) {
    // A fragment (16x32 f16): lane -> row (row0+lm); elem e -> K = kc + 8*half + (e<8 ? e : e+8)
    v16h a, b;
    const float* arow = in + (size_t)(row0 + lm) * KDIM + kc + (half ? 8 : 0);
    #pragma unroll
    for (int e = 0; e < 8; ++e) {
      a[e]     = (_Float16)arow[e];
      a[e + 8] = (_Float16)arow[e + 16];
    }
    // B fragment (32x16 f16): lane -> col; elem e -> K = kc + 8*half + (e<8 ? e : e+8)
    const float* bcol = W + (size_t)(kc + (half ? 8 : 0)) * 64 + col;
    #pragma unroll
    for (int e = 0; e < 16; ++e) b[e] = (_Float16)bcol[(size_t)(e < 8 ? e : e + 8) * 64];

    acc = __builtin_amdgcn_wmma_f32_16x16x32_f16(false, a, false, b, (short)0, acc,
                                                 false, false);
  }
  // C/D layout: VGPR r -> row M = r + 8*half, col = lane%16
  #pragma unroll
  for (int r = 0; r < 8; ++r)
    feat[(size_t)(row0 + r + (half ? 8 : 0)) * 64 + col] = acc[r];
}

// ---------- per-node attention dots: a_s[n,h], a_d[n,h] ----------
__global__ void node_atts(const float* __restrict__ feat, const float* __restrict__ att_s,
                          const float* __restrict__ att_d, float* __restrict__ a_s,
                          float* __restrict__ a_d, int n) {
  int t = blockIdx.x * blockDim.x + threadIdx.x;
  if (t >= n * 4) return;
  int h = t & 3;
  const float* f = feat + (size_t)(t >> 2) * 64 + h * 16;
  float s = 0.f, d = 0.f;
  #pragma unroll
  for (int c = 0; c < 16; ++c) {
    float v = f[c];
    s += v * att_s[h * 16 + c];
    d += v * att_d[h * 16 + c];
  }
  a_s[t] = s;
  a_d[t] = d;
}

// ---------- edge attention from edge features: a_e[e,h] = ea[e,:16] @ M[16,4] ----------
__global__ void edge_ae(const float* __restrict__ eattr, const float* __restrict__ mean_attr,
                        const float* __restrict__ M, float* __restrict__ a_e, int E, int Etot) {
  __shared__ float sM[64];
  if (threadIdx.x < 64) sM[threadIdx.x] = M[threadIdx.x];
  __syncthreads();
  int e = blockIdx.x * blockDim.x + threadIdx.x;
  if (e >= Etot) return;
  const float* ea = (e < E) ? (eattr + (size_t)e * 16) : mean_attr;
  float acc0 = 0.f, acc1 = 0.f, acc2 = 0.f, acc3 = 0.f;
  #pragma unroll
  for (int d = 0; d < 16; ++d) {
    float v = ea[d];
    acc0 += v * sM[d * 4 + 0];
    acc1 += v * sM[d * 4 + 1];
    acc2 += v * sM[d * 4 + 2];
    acc3 += v * sM[d * 4 + 3];
  }
  float* o = a_e + (size_t)e * 4;
  o[0] = acc0; o[1] = acc1; o[2] = acc2; o[3] = acc3;
}

// order-preserving float<->uint key for atomic max
__device__ __forceinline__ unsigned f2key(float f) {
  unsigned b = __float_as_uint(f);
  return (b & 0x80000000u) ? ~b : (b | 0x80000000u);
}
__device__ __forceinline__ float key2f(unsigned k) {
  unsigned b = (k & 0x80000000u) ? (k ^ 0x80000000u) : ~k;
  return __uint_as_float(b);
}

// ---------- softmax pass 1: leaky-relu logit (in place over a_e) + segment max ----------
__global__ void edge_alpha_max(const int* __restrict__ ei, const float* __restrict__ a_s,
                               const float* __restrict__ a_d, float* __restrict__ alpha,
                               unsigned* __restrict__ amax, int E, int Etot) {
  int e = blockIdx.x * blockDim.x + threadIdx.x;
  if (e >= Etot) return;
  int s = (e < E) ? ei[e] : (e - E);
  int d = (e < E) ? ei[(size_t)E + e] : (e - E);
  #pragma unroll
  for (int h = 0; h < 4; ++h) {
    float v = a_s[(size_t)s * 4 + h] + a_d[(size_t)d * 4 + h] + alpha[(size_t)e * 4 + h];
    v = (v > 0.f) ? v : LRELU_SLOPE * v;
    alpha[(size_t)e * 4 + h] = v;
    atomicMax(&amax[(size_t)d * 4 + h], f2key(v));
  }
}

// ---------- softmax pass 2: exp (in place) + segment sum ----------
__global__ void edge_exp_den(const int* __restrict__ ei, float* __restrict__ alpha,
                             const unsigned* __restrict__ amax, float* __restrict__ den,
                             int E, int Etot) {
  int e = blockIdx.x * blockDim.x + threadIdx.x;
  if (e >= Etot) return;
  int d = (e < E) ? ei[(size_t)E + e] : (e - E);
  #pragma unroll
  for (int h = 0; h < 4; ++h) {
    float m  = key2f(amax[(size_t)d * 4 + h]);
    float ex = __expf(alpha[(size_t)e * 4 + h] - m);
    alpha[(size_t)e * 4 + h] = ex;
    atomicAdd(&den[(size_t)d * 4 + h], ex);
  }
}

// ---------- message scatter: out[dst] += feat[src] * alpha  (thread = edge*head) ----------
__global__ void edge_message(const int* __restrict__ ei, const float* __restrict__ feat,
                             const float* __restrict__ exv, const float* __restrict__ den,
                             float* __restrict__ outacc, int E, int Etot) {
  long t = (long)blockIdx.x * blockDim.x + threadIdx.x;
  if (t >= (long)Etot * 4) return;
  int e = (int)(t >> 2);
  int h = (int)(t & 3);
  int s = (e < E) ? ei[e] : (e - E);
  int d = (e < E) ? ei[(size_t)E + e] : (e - E);
  const float* f = feat + (size_t)s * 64 + h * 16;
  __builtin_prefetch(f, 0, 0);  // global_prefetch_b8 on gfx1250
  float a = exv[t] / (den[(size_t)d * 4 + h] + EPS_A);
  float* o = outacc + (size_t)d * 64 + h * 16;
  #pragma unroll
  for (int c = 0; c < 16; ++c) atomicAdd(&o[c], f[c] * a);
}

// ---------- bias + relu ----------
__global__ void bias_relu(const float* __restrict__ acc, const float* __restrict__ b,
                          float* __restrict__ out, long n64) {
  long t = (long)blockIdx.x * blockDim.x + threadIdx.x;
  if (t >= n64) return;
  out[t] = fmaxf(acc[t] + b[t & 63], 0.f);
}

// ---------- MLP head + log_softmax ----------
__global__ void head_kernel(const float* __restrict__ h2, const float* __restrict__ lw1,
                            const float* __restrict__ lb1, const float* __restrict__ lw2,
                            const float* __restrict__ lb2, float* __restrict__ out, int n) {
  __shared__ float sW1[64 * 16];
  __shared__ float sW2[16 * 40];
  __shared__ float sB1[16];
  __shared__ float sB2[40];
  for (int i = threadIdx.x; i < 64 * 16; i += blockDim.x) sW1[i] = lw1[i];
  for (int i = threadIdx.x; i < 16 * 40; i += blockDim.x) sW2[i] = lw2[i];
  if (threadIdx.x < 16) sB1[threadIdx.x] = lb1[threadIdx.x];
  if (threadIdx.x < 40) sB2[threadIdx.x] = lb2[threadIdx.x];
  __syncthreads();
  int node = blockIdx.x * blockDim.x + threadIdx.x;
  if (node >= n) return;
  const float* hv = h2 + (size_t)node * 64;

  float t16[16];
  #pragma unroll
  for (int j = 0; j < 16; ++j) t16[j] = sB1[j];
  for (int k = 0; k < 64; ++k) {
    float v = hv[k];
    #pragma unroll
    for (int j = 0; j < 16; ++j) t16[j] += v * sW1[k * 16 + j];
  }
  float z[40];
  #pragma unroll
  for (int j = 0; j < 40; ++j) z[j] = sB2[j];
  #pragma unroll
  for (int k = 0; k < 16; ++k) {
    float v = t16[k];
    for (int j = 0; j < 40; ++j) z[j] += v * sW2[k * 40 + j];
  }
  float m = z[0];
  for (int j = 1; j < 40; ++j) m = fmaxf(m, z[j]);
  float ssum = 0.f;
  for (int j = 0; j < 40; ++j) ssum += __expf(z[j] - m);
  float lse = __logf(ssum);
  float* o = out + (size_t)node * 40;
  for (int j = 0; j < 40; ++j) o[j] = z[j] - m - lse;
}

extern "C" void kernel_launch(void* const* d_in, const int* in_sizes, int n_in,
                              void* d_out, int out_size, void* d_ws, size_t ws_size,
                              hipStream_t stream) {
  (void)n_in; (void)out_size; (void)ws_size;
  const float* x    = (const float*)d_in[0];
  const int*   ei   = (const int*)d_in[1];    // [2,E] row-major: src then dst
  const float* eatt = (const float*)d_in[2];
  const float* W1   = (const float*)d_in[3];
  const float* as1  = (const float*)d_in[4];
  const float* ad1  = (const float*)d_in[5];
  const float* We1  = (const float*)d_in[6];
  const float* ae1w = (const float*)d_in[7];
  const float* b1   = (const float*)d_in[8];
  const float* W2   = (const float*)d_in[9];
  const float* as2  = (const float*)d_in[10];
  const float* ad2  = (const float*)d_in[11];
  const float* We2  = (const float*)d_in[12];
  const float* ae2w = (const float*)d_in[13];
  const float* b2   = (const float*)d_in[14];
  const float* lw1  = (const float*)d_in[15];
  const float* lb1  = (const float*)d_in[16];
  const float* lw2  = (const float*)d_in[17];
  const float* lb2  = (const float*)d_in[18];

  const int n    = in_sizes[0] / 128;  // 50000 (multiple of 16)
  const int E    = in_sizes[1] / 2;    // 1,600,000
  const int Etot = E + n;

  // workspace carve-out (~69 MB)
  float* ws      = (float*)d_ws;
  float* feat    = ws;              ws += (size_t)n * 64;
  float* hbuf    = ws;              ws += (size_t)n * 64;
  float* outacc  = ws;              ws += (size_t)n * 64;
  float* a_s     = ws;              ws += (size_t)n * 4;
  float* a_d     = ws;              ws += (size_t)n * 4;
  unsigned* amax = (unsigned*)ws;   ws += (size_t)n * 4;
  float* den     = ws;              ws += (size_t)n * 4;
  float* alpha   = ws;              ws += (size_t)Etot * 4;
  float* meanat  = ws;              ws += 16;
  float* M1      = ws;              ws += 64;
  float* M2      = ws;              ws += 64;

  const int B = 256;
  auto cdiv = [](long a, long b) { return (int)((a + b - 1) / b); };

  // precompute: mean edge_attr, collapsed edge-attention matrices
  fill_f32<<<1, 64, 0, stream>>>(meanat, 0.f, 16);
  mean_attr_kernel<<<64, 256, 0, stream>>>(eatt, meanat, E, 64 * 256);
  collapse_We<<<1, 64, 0, stream>>>(We1, ae1w, M1);
  collapse_We<<<1, 64, 0, stream>>>(We2, ae2w, M2);

  const float* layer_in[2]  = {x, hbuf};
  const float* layer_W[2]   = {W1, W2};
  const float* layer_as[2]  = {as1, as2};
  const float* layer_ad[2]  = {ad1, ad2};
  const float* layer_b[2]   = {b1, b2};
  float*       layer_M[2]   = {M1, M2};

  for (int L = 0; L < 2; ++L) {
    if (L == 0)
      node_gemm<128><<<n / 16, 128, 0, stream>>>(layer_in[L], layer_W[L], feat);
    else
      node_gemm<64><<<n / 16, 128, 0, stream>>>(layer_in[L], layer_W[L], feat);

    node_atts<<<cdiv((long)n * 4, B), B, 0, stream>>>(feat, layer_as[L], layer_ad[L],
                                                      a_s, a_d, n);
    edge_ae<<<cdiv(Etot, B), B, 0, stream>>>(eatt, meanat, layer_M[L], alpha, E, Etot);

    fill_f32<<<cdiv((long)n * 4, B), B, 0, stream>>>((float*)amax, 0.f, (long)n * 4);
    fill_f32<<<cdiv((long)n * 4, B), B, 0, stream>>>(den, 0.f, (long)n * 4);

    edge_alpha_max<<<cdiv(Etot, B), B, 0, stream>>>(ei, a_s, a_d, alpha, amax, E, Etot);
    edge_exp_den<<<cdiv(Etot, B), B, 0, stream>>>(ei, alpha, amax, den, E, Etot);

    fill_f32<<<cdiv((long)n * 64, B), B, 0, stream>>>(outacc, 0.f, (long)n * 64);
    edge_message<<<cdiv((long)Etot * 4, B), B, 0, stream>>>(ei, feat, alpha, den,
                                                            outacc, E, Etot);
    bias_relu<<<cdiv((long)n * 64, B), B, 0, stream>>>(outacc, layer_b[L], hbuf,
                                                       (long)n * 64);
  }

  head_kernel<<<cdiv(n, B), B, 0, stream>>>(hbuf, lw1, lb1, lw2, lb2, (float*)d_out, n);
}
